// RPN_63187558858814
// MI455X (gfx1250) — matrix operand order repaired and checked by reading, not compiled, not run on hardware
//
#include <hip/hip_runtime.h>
#include <hip/hip_bf16.h>
#include <math.h>

typedef __attribute__((ext_vector_type(16))) __bf16 v16bf_t;
typedef __attribute__((ext_vector_type(8)))  __bf16 v8bf_t;
typedef __attribute__((ext_vector_type(8)))  float  v8f_t;

namespace {
constexpr int Bn  = 2;
constexpr int Cin = 1024;
constexpr int H   = 50;
constexpr int W   = 76;
constexpr int OC  = 512;
constexpr int PB  = H * W;        // 3800 pixels per batch
constexpr int P   = Bn * PB;      // 7600 pixels total
constexpr int HP  = H + 2;        // padded
constexpr int WP  = W + 2;
constexpr int NA  = PB * 9;       // 34200 anchors per batch
constexpr int NS  = 65536;        // padded bitonic size
constexpr int CHUNK = 2048;       // LDS bitonic chunk
constexpr int PRE = 2000;
constexpr int POST = 300;
constexpr float NMS_T = 0.7f;
}

// ---------- 1) fp32 NCHW -> bf16 padded NHWC ----------
__global__ __launch_bounds__(256) void k_pad_nhwc(const float* __restrict__ src,
                                                  __bf16* __restrict__ dst) {
  int id = blockIdx.x * 256 + threadIdx.x;
  const int total = Bn * HP * WP * Cin;
  if (id >= total) return;
  int ic = id & (Cin - 1);
  int t  = id >> 10;
  int xp = t % WP; t /= WP;
  int yp = t % HP;
  int b  = t / HP;
  int y = yp - 1, x = xp - 1;
  float v = 0.f;
  if ((unsigned)y < (unsigned)H && (unsigned)x < (unsigned)W)
    v = src[((b * Cin + ic) * H + y) * W + x];
  dst[id] = (__bf16)v;
}

// ---------- 2) conv_w fp32 [oc][ic][ky][kx] -> bf16 [kykx][oc][ic] ----------
__global__ __launch_bounds__(256) void k_pack_w(const float* __restrict__ src,
                                                __bf16* __restrict__ dst) {
  int id = blockIdx.x * 256 + threadIdx.x;
  const int total = 9 * OC * Cin;
  if (id >= total) return;
  int ic = id & (Cin - 1);
  int t  = id >> 10;
  int oc = t % OC;
  int kk = t / OC;
  int ky = kk / 3, kx = kk % 3;
  dst[id] = (__bf16)src[((oc * Cin + ic) * 3 + ky) * 3 + kx];
}

// ---------- 3) 3x3 conv as implicit GEMM via WMMA bf16 ----------
// block: 8 waves = 64 oc x 128 px; wave: 16 oc x 64 px (4 accumulators)
__global__ __launch_bounds__(256) void k_conv3_wmma(
    const __bf16* __restrict__ xpad, const __bf16* __restrict__ wpk,
    const float* __restrict__ bias, __bf16* __restrict__ xrelu) {
  const int lane  = threadIdx.x & 31;
  const int wid   = threadIdx.x >> 5;
  const int ocSub = wid & 3;        // 16-oc subtile within block
  const int pxg   = wid >> 2;       // 64-px group within block
  const int laneM = lane & 15;
  const int laneH = lane >> 4;      // 0 or 1
  const int oc16  = blockIdx.y * 64 + ocSub * 16;

  // A-operand: row oc16+laneM; contiguous K order used here (same in B): lanes 0-15 hold K 0-15,
  // lanes 16-31 hold K 16-31  -> chunk base kb = laneH*16
  const int kb   = laneH * 16;
  const int aRow = (oc16 + laneM) * Cin;

  // B-operand: N = laneM pixel, K contiguous in NHWC; lane half selects +16 K
  int bOff[4];
  int pixq[4];
#pragma unroll
  for (int q = 0; q < 4; ++q) {
    int p = blockIdx.x * 128 + pxg * 64 + q * 16 + laneM;
    pixq[q] = p;
    int pc = p < P ? p : P - 1;               // clamp (no divergence pre-WMMA)
    int b = pc / PB, r = pc % PB;
    int y = r / W, x = r % W;
    bOff[q] = ((b * HP + y) * WP + x) * Cin + laneH * 16;
  }

  v8f_t acc[4] = {};

  for (int kk = 0; kk < 9; ++kk) {
    const int ky = kk / 3, kx = kk % 3;
    const __bf16* aSlab = wpk + kk * (OC * Cin) + aRow;
    const int bShift = (ky * WP + kx) * Cin;
#pragma unroll 4
    for (int icb = 0; icb < Cin; icb += 32) {
      v8bf_t alo = *(const v8bf_t*)(aSlab + icb + kb);
      v8bf_t ahi = *(const v8bf_t*)(aSlab + icb + kb + 8);
      v16bf_t av;
#pragma unroll
      for (int t = 0; t < 8; ++t) { av[t] = alo[t]; av[8 + t] = ahi[t]; }
#pragma unroll
      for (int q = 0; q < 4; ++q) {
        v16bf_t bv = *(const v16bf_t*)(xpad + bOff[q] + bShift + icb);
        acc[q] = __builtin_amdgcn_wmma_f32_16x16x32_bf16(
            false, av, false, bv, (short)0, acc[q], false, false);
      }
    }
  }

  // C/D layout: M = laneH*8 + r, N = laneM  -> 8 consecutive oc per lane
  const int ocB = oc16 + laneH * 8;
#pragma unroll
  for (int q = 0; q < 4; ++q) {
    if (pixq[q] >= P) continue;
    v8bf_t outv;
#pragma unroll
    for (int r = 0; r < 8; ++r) {
      float f = acc[q][r] + bias[ocB + r];
      f = f > 0.f ? f : 0.f;
      outv[r] = (__bf16)f;
    }
    *(v8bf_t*)(xrelu + (size_t)pixq[q] * OC + ocB) = outv;
  }
}

// ---------- 4) 1x1 convs (cls 18 + reg 36) ----------
__global__ __launch_bounds__(256) void k_conv1(
    const __bf16* __restrict__ xrelu,
    const float* __restrict__ cls_w, const float* __restrict__ cls_b,
    const float* __restrict__ reg_w, const float* __restrict__ reg_b,
    float* __restrict__ cls, float* __restrict__ reg) {
  int id = blockIdx.x * 256 + threadIdx.x;
  if (id >= P * 54) return;
  int oc = id % 54;
  int p  = id / 54;
  const __bf16* xr = xrelu + (size_t)p * OC;
  const float* wrow;
  float acc;
  if (oc < 18) { wrow = cls_w + oc * OC;        acc = cls_b[oc]; }
  else         { wrow = reg_w + (oc - 18) * OC; acc = reg_b[oc - 18]; }
#pragma unroll 8
  for (int k = 0; k < OC; ++k) acc += (float)xr[k] * wrow[k];
  if (oc < 18) cls[(size_t)p * 18 + oc]        = acc;
  else         reg[(size_t)p * 36 + (oc - 18)] = acc;
}

// ---------- 5) softmax fg score, anchor gen, decode, clip, sort keys ----------
__global__ __launch_bounds__(256) void k_score(
    const float* __restrict__ cls, const float* __restrict__ reg,
    const float* __restrict__ iminfo,
    float* __restrict__ boxes, unsigned long long* __restrict__ keys) {
  int id = blockIdx.x * 256 + threadIdx.x;
  if (id >= Bn * NS) return;
  int b = id >> 16;
  int r = id & (NS - 1);
  if (r >= NA) {  // padding: sorts to the end
    keys[id] = (0xFFFFFFFFull << 32) | (unsigned)r;
    return;
  }
  int pl = r / 9, a = r % 9;
  int y = pl / W, x = pl % W;
  int pg = b * PB + pl;
  float s0 = cls[(size_t)pg * 18 + a];
  float s1 = cls[(size_t)pg * 18 + 9 + a];
  float fg = 1.f / (1.f + expf(s0 - s1));
  // base anchor a = ratio_idx*3 + scale_idx (base_size=16)
  const float ratios[3] = {0.5f, 1.f, 2.f};
  const float scales[3] = {8.f, 16.f, 32.f};
  int ri = a / 3, si = a % 3;
  float ws = rintf(sqrtf(256.f / ratios[ri]));
  float hs = rintf(ws * ratios[ri]);
  float sw = ws * scales[si], sh = hs * scales[si];
  float ax1 = 7.5f - 0.5f * (sw - 1.f) + x * 16.f;
  float ay1 = 7.5f - 0.5f * (sh - 1.f) + y * 16.f;
  float ax2 = 7.5f + 0.5f * (sw - 1.f) + x * 16.f;
  float ay2 = 7.5f + 0.5f * (sh - 1.f) + y * 16.f;
  float aw = ax2 - ax1 + 1.f, ah = ay2 - ay1 + 1.f;
  float acx = ax1 + 0.5f * aw, acy = ay1 + 0.5f * ah;
  const float* d = reg + (size_t)pg * 36 + a * 4;
  float pcx = d[0] * aw + acx;
  float pcy = d[1] * ah + acy;
  float pw  = expf(d[2]) * aw;
  float ph  = expf(d[3]) * ah;
  float imh = iminfo[b * 3 + 0] - 1.f;
  float imw = iminfo[b * 3 + 1] - 1.f;
  float x1 = fminf(fmaxf(pcx - 0.5f * pw, 0.f), imw);
  float y1 = fminf(fmaxf(pcy - 0.5f * ph, 0.f), imh);
  float x2 = fminf(fmaxf(pcx + 0.5f * pw, 0.f), imw);
  float y2 = fminf(fmaxf(pcy + 0.5f * ph, 0.f), imh);
  float* bx = boxes + ((size_t)b * NA + r) * 4;
  bx[0] = x1; bx[1] = y1; bx[2] = x2; bx[3] = y2;
  unsigned u = __float_as_uint(fg);
  u = (u >> 31) ? ~u : (u | 0x80000000u);  // monotone ascending map
  unsigned kd = ~u;                         // descending by score
  keys[id] = ((unsigned long long)kd << 32) | (unsigned)r;  // ties: lower idx first
}

// ---------- 6a) bitonic: full local sort of 2048-key chunks (k = 2..2048) ----------
__global__ __launch_bounds__(1024) void k_bitonic_local(unsigned long long* __restrict__ keys) {
  __shared__ unsigned long long s[CHUNK];
  const int chunk = blockIdx.x;
  unsigned long long* g = keys + (size_t)chunk * CHUNK;
  const int tid = threadIdx.x;
  s[tid] = g[tid];
  s[tid + 1024] = g[tid + 1024];
  __syncthreads();
  const int gbase = (chunk * CHUNK) & (NS - 1);  // within-batch offset
  for (int k = 2; k <= CHUNK; k <<= 1) {
    for (int j = k >> 1; j > 0; j >>= 1) {
      int i   = ((tid & ~(j - 1)) << 1) | (tid & (j - 1));
      int ixj = i | j;
      bool up = (((gbase | i) & k) == 0);
      unsigned long long a = s[i], c = s[ixj];
      if ((a > c) == up) { s[i] = c; s[ixj] = a; }
      __syncthreads();
    }
  }
  g[tid] = s[tid];
  g[tid + 1024] = s[tid + 1024];
}

// ---------- 6b) bitonic: global pass for strides j >= CHUNK ----------
__global__ __launch_bounds__(256) void k_bitonic(unsigned long long* __restrict__ keys,
                                                 int j, int k) {
  int id = blockIdx.x * 256 + threadIdx.x;
  if (id >= Bn * NS) return;
  int base = (id >> 16) << 16;
  int i = id & (NS - 1);
  int ixj = i ^ j;
  if (ixj <= i) return;
  unsigned long long a = keys[base + i];
  unsigned long long c = keys[base + ixj];
  bool up = ((i & k) == 0);
  if ((a > c) == up) { keys[base + i] = c; keys[base + ixj] = a; }
}

// ---------- 6c) bitonic: local merge (j = 1024..1) for outer stage k ----------
__global__ __launch_bounds__(1024) void k_bitonic_merge_local(
    unsigned long long* __restrict__ keys, int k) {
  __shared__ unsigned long long s[CHUNK];
  const int chunk = blockIdx.x;
  unsigned long long* g = keys + (size_t)chunk * CHUNK;
  const int tid = threadIdx.x;
  s[tid] = g[tid];
  s[tid + 1024] = g[tid + 1024];
  __syncthreads();
  const int gbase = (chunk * CHUNK) & (NS - 1);
  const bool up = ((gbase & k) == 0);  // uniform for the whole chunk (k > CHUNK)
  for (int j = CHUNK >> 1; j > 0; j >>= 1) {
    int i   = ((tid & ~(j - 1)) << 1) | (tid & (j - 1));
    int ixj = i | j;
    unsigned long long a = s[i], c = s[ixj];
    if ((a > c) == up) { s[i] = c; s[ixj] = a; }
    __syncthreads();
  }
  g[tid] = s[tid];
  g[tid + 1024] = s[tid + 1024];
}

// ---------- 7) gather top-2000 boxes in score order ----------
__global__ __launch_bounds__(256) void k_gather(const unsigned long long* __restrict__ keys,
                                                const float* __restrict__ boxes,
                                                float* __restrict__ topb) {
  int id = blockIdx.x * 256 + threadIdx.x;
  if (id >= Bn * PRE) return;
  int b = id / PRE, t = id % PRE;
  unsigned idx = (unsigned)(keys[(size_t)b * NS + t] & 0xFFFFFFFFu);
  const float* s = boxes + ((size_t)b * NA + idx) * 4;
  float* dd = topb + (size_t)id * 4;
  dd[0] = s[0]; dd[1] = s[1]; dd[2] = s[2]; dd[3] = s[3];
}

// ---------- 8a) suppression bitmask: mask[b][i][w] bit e set if IoU(i, w*64+e)>T, j>i ----------
__global__ __launch_bounds__(256) void k_iou_mask(const float* __restrict__ topb,
                                                  unsigned long long* __restrict__ mask) {
  int id = blockIdx.x * 256 + threadIdx.x;
  if (id >= Bn * PRE * 32) return;
  int w = id & 31;
  int t = id >> 5;           // b*PRE + i
  int b = t / PRE, i = t % PRE;
  const float4* tb = (const float4*)topb + (size_t)b * PRE;
  float4 bi = tb[i];
  float ai = (bi.z - bi.x + 1.f) * (bi.w - bi.y + 1.f);
  unsigned long long m = 0ull;
  int j0 = w * 64;
  for (int e = 0; e < 64; ++e) {
    int j = j0 + e;
    if (j <= i || j >= PRE) continue;
    float4 bj = tb[j];
    float aj = (bj.z - bj.x + 1.f) * (bj.w - bj.y + 1.f);
    float xx1 = fmaxf(bi.x, bj.x), yy1 = fmaxf(bi.y, bj.y);
    float xx2 = fminf(bi.z, bj.z), yy2 = fminf(bi.w, bj.w);
    float iw = fmaxf(xx2 - xx1 + 1.f, 0.f);
    float ih = fmaxf(yy2 - yy1 + 1.f, 0.f);
    float inter = iw * ih;
    float iou = inter / (ai + aj - inter);
    if (iou > NMS_T) m |= (1ull << e);
  }
  mask[((size_t)b * PRE + i) * 32 + w] = m;
}

// ---------- 8b) sequential NMS scan: one wave32 per batch, no barriers ----------
// Lane l owns keep-bits for boxes j in [64l, 64l+63]; mask rows prefetched 8 deep.
__global__ __launch_bounds__(32) void k_nms_wave(const unsigned long long* __restrict__ mask,
                                                 unsigned* __restrict__ keep) {
  const int b = blockIdx.x;
  const int lane = threadIdx.x;
  unsigned long long keepw = (lane == 31) ? 0xFFFFull : ~0ull;  // trim j >= 2000
  const unsigned long long* mrow = mask + (size_t)b * PRE * 32;
  unsigned long long pend[8];
#pragma unroll
  for (int q = 0; q < 8; ++q) pend[q] = mrow[(size_t)q * 32 + lane];
#pragma unroll 8
  for (int i = 0; i < PRE; ++i) {
    unsigned long long row = pend[i & 7];
    unsigned bit = (unsigned)((keepw >> (i & 63)) & 1ull);
    bit = __shfl(bit, i >> 6, 32);       // broadcast keep-state of box i
    if (bit) keepw &= ~row;
    int nx = i + 8;
    if (nx < PRE) pend[i & 7] = mrow[(size_t)nx * 32 + lane];
  }
#pragma unroll
  for (int e = 0; e < 64; ++e) {
    int t = lane * 64 + e;
    if (t < PRE) keep[b * PRE + t] = (unsigned)((keepw >> e) & 1ull);
  }
}

// ---------- 9) compact first 300 kept boxes -> rois ----------
__global__ void k_final(const float* __restrict__ topb, const unsigned* __restrict__ keep,
                        float* __restrict__ out) {
  int b = blockIdx.x;
  if (threadIdx.x != 0) return;
  int cnt = 0;
  for (int t = 0; t < PRE && cnt < POST; ++t) {
    if (keep[b * PRE + t]) {
      const float* s = topb + ((size_t)b * PRE + t) * 4;
      float* dd = out + ((size_t)b * POST + cnt) * 5;
      dd[0] = (float)b; dd[1] = s[0]; dd[2] = s[1]; dd[3] = s[2]; dd[4] = s[3];
      ++cnt;
    }
  }
  for (int r2 = cnt; r2 < POST; ++r2) {
    float* dd = out + ((size_t)b * POST + r2) * 5;
    dd[0] = 0.f; dd[1] = 0.f; dd[2] = 0.f; dd[3] = 0.f; dd[4] = 0.f;
  }
}

extern "C" void kernel_launch(void* const* d_in, const int* in_sizes, int n_in,
                              void* d_out, int out_size, void* d_ws, size_t ws_size,
                              hipStream_t stream) {
  const float* base_feat = (const float*)d_in[0];
  const float* iminfo    = (const float*)d_in[1];
  const float* conv_w    = (const float*)d_in[2];
  const float* conv_b    = (const float*)d_in[3];
  const float* cls_w     = (const float*)d_in[4];
  const float* cls_b     = (const float*)d_in[5];
  const float* reg_w     = (const float*)d_in[6];
  const float* reg_b     = (const float*)d_in[7];
  float* out = (float*)d_out;

  char* wsp = (char*)d_ws;
  auto carve = [&](size_t bytes) -> void* {
    void* p = (void*)wsp;
    wsp += (bytes + 255) & ~(size_t)255;
    return p;
  };
  __bf16* xpad  = (__bf16*)carve((size_t)Bn * HP * WP * Cin * 2);
  __bf16* wpk   = (__bf16*)carve((size_t)9 * OC * Cin * 2);
  __bf16* xrelu = (__bf16*)carve((size_t)P * OC * 2);
  float*  cls   = (float*)carve((size_t)P * 18 * 4);
  float*  reg   = (float*)carve((size_t)P * 36 * 4);
  float*  boxes = (float*)carve((size_t)Bn * NA * 4 * 4);
  unsigned long long* keys = (unsigned long long*)carve((size_t)Bn * NS * 8);
  float*  topb  = (float*)carve((size_t)Bn * PRE * 4 * 4);
  unsigned long long* mask = (unsigned long long*)carve((size_t)Bn * PRE * 32 * 8);
  unsigned* keep = (unsigned*)carve((size_t)Bn * PRE * 4);

  const int tPad = Bn * HP * WP * Cin;
  k_pad_nhwc<<<(tPad + 255) / 256, 256, 0, stream>>>(base_feat, xpad);
  const int tW = 9 * OC * Cin;
  k_pack_w<<<(tW + 255) / 256, 256, 0, stream>>>(conv_w, wpk);

  dim3 g3((P + 127) / 128, OC / 64);
  k_conv3_wmma<<<g3, 256, 0, stream>>>(xpad, wpk, conv_b, xrelu);

  k_conv1<<<(P * 54 + 255) / 256, 256, 0, stream>>>(xrelu, cls_w, cls_b, reg_w, reg_b,
                                                    cls, reg);
  k_score<<<(Bn * NS) / 256, 256, 0, stream>>>(cls, reg, iminfo, boxes, keys);

  // bitonic sort, 21 launches total: LDS full sort (k<=2048), then per outer
  // stage the j>=2048 global passes + one LDS merge for j<=1024.
  const int nChunks = Bn * NS / CHUNK;
  k_bitonic_local<<<nChunks, 1024, 0, stream>>>(keys);
  for (int k = CHUNK * 2; k <= NS; k <<= 1) {
    for (int j = k >> 1; j >= CHUNK; j >>= 1)
      k_bitonic<<<(Bn * NS) / 256, 256, 0, stream>>>(keys, j, k);
    k_bitonic_merge_local<<<nChunks, 1024, 0, stream>>>(keys, k);
  }

  k_gather<<<(Bn * PRE + 255) / 256, 256, 0, stream>>>(keys, boxes, topb);
  k_iou_mask<<<(Bn * PRE * 32 + 255) / 256, 256, 0, stream>>>(topb, mask);
  k_nms_wave<<<Bn, 32, 0, stream>>>(mask, keep);
  k_final<<<Bn, 1, 0, stream>>>(topb, keep, out);

  (void)in_sizes; (void)n_in; (void)out_size; (void)ws_size;
}
